// ElementReadoutMLP_56959856280416
// MI455X (gfx1250) — matrix operand order repaired and checked by reading, not compiled, not run on hardware
//
#include <hip/hip_runtime.h>

// ---------------------------------------------------------------------------
// ElementReadoutMLP for gfx1250 (MI455X), wave32 + WMMA f16 split-precision.
//
//   out[a] = W2[e]^T . silu(W1[e]^T x[a] + b1[e]) + b2[e],  e = species[a]
//
// Pass 1: bucket atoms by species (count / scan / scatter) into d_ws.
// Pass 2: per (species, 128-atom tile) block of 512 threads (16 wave32):
//         waves arranged 8 (M) x 2 (N-half). Each wave owns 16 rows x 128
//         cols = 8 WMMA C-tiles (64 accum VGPRs -> no vgpr_msb, good
//         occupancy). K loop stages X rows (gathered) and W1[e] into LDS as
//         f16 hi/lo planes; 3x V_WMMA_F32_16X16X32_F16 per tile gives
//         ~fp32 accuracy at the f16 matrix rate. SiLU + the [256 x 1]
//         second layer are fused in-registers, with one 256-float LDS
//         combine across the two N-halves.
// ---------------------------------------------------------------------------

typedef __attribute__((ext_vector_type(16))) _Float16 v16h;
typedef __attribute__((ext_vector_type(8)))  float    v8f;

#define NDIM   256
#define MTILE  128      // atoms per block
#define KC     32       // K chunk (one f16 WMMA depth)
#define LDA    34       // LDS stride (halves): 17 banks -> conflict-free frags
#define LDB    34

union HFrag {
    v16h v;
    unsigned int u[8];
};

// ---------------------------- bucketing kernels ----------------------------

__global__ void erm_init(int* ws) {
    if (threadIdx.x < 48) ws[threadIdx.x] = 0;   // counts[16]+cursor[16]+offsets[16]
}

__global__ void erm_count(const int* __restrict__ species, int* counts, int nAtoms) {
    int i = blockIdx.x * blockDim.x + threadIdx.x;
    if (i < nAtoms) atomicAdd(&counts[species[i]], 1);
}

__global__ void erm_scan(int* ws, int nElem) {
    if (threadIdx.x == 0) {
        int* counts  = ws;
        int* cursor  = ws + 16;
        int* offsets = ws + 32;
        int off = 0;
        for (int e = 0; e < nElem; ++e) {
            offsets[e] = off;
            cursor[e]  = off;
            off += counts[e];
        }
    }
}

__global__ void erm_scatter(const int* __restrict__ species, int* cursor,
                            int* sortedIdx, int nAtoms) {
    int i = blockIdx.x * blockDim.x + threadIdx.x;
    if (i < nAtoms) {
        int pos = atomicAdd(&cursor[species[i]], 1);
        sortedIdx[pos] = i;
    }
}

// ------------------------------ fused MLP ----------------------------------

__global__ __launch_bounds__(512) void erm_mlp(
    const float* __restrict__ X,        // [nAtoms, 256]
    const float* __restrict__ W1,       // [nElem, 256, 256]
    const float* __restrict__ b1,       // [nElem, 256]
    const float* __restrict__ W2,       // [nElem, 256, 1]
    const float* __restrict__ b2,       // [nElem, 1]
    const int*   __restrict__ counts,   // [16]
    const int*   __restrict__ offsets,  // [16]
    const int*   __restrict__ sortedIdx,// [nAtoms]
    float*       __restrict__ out)      // [nAtoms]
{
    __shared__ _Float16 sAh[MTILE * LDA];
    __shared__ _Float16 sAl[MTILE * LDA];
    __shared__ _Float16 sBh[NDIM * LDB];   // transposed: [n][k]
    __shared__ _Float16 sBl[NDIM * LDB];
    __shared__ float    sB1[NDIM];
    __shared__ float    sW2[NDIM];
    __shared__ float    sPart[2 * MTILE]; // N-half partial row sums

    const int e      = blockIdx.y;
    const int cnt    = counts[e];
    const int mStart = blockIdx.x * MTILE;
    if (mStart >= cnt) return;             // uniform exit: EXEC all-ones below

    const int tid  = threadIdx.x;
    const int lane = tid & 31;
    const int wid  = tid >> 5;             // 16 waves
    const int mw   = wid & 7;              // M wave: rows [16*mw, 16*mw+16)
    const int nh   = wid >> 3;             // N half: cols [128*nh, 128*nh+128)
    const int nl   = lane & 15;
    const int lh   = lane >> 4;            // lane half (0/1)
    const int off  = offsets[e];

    if (tid < NDIM) {                      // per-element bias + layer-2 weights
        sB1[tid] = b1[(size_t)e * NDIM + tid];
        sW2[tid] = W2[(size_t)e * NDIM + tid];
    }

    v8f c[8];
    #pragma unroll
    for (int t = 0; t < 8; ++t) c[t] = (v8f){0.f,0.f,0.f,0.f,0.f,0.f,0.f,0.f};

    // lane-invariant LDS fragment bases (halves), hoisted out of the K loop
    const int aRowBase = (16 * mw + nl) * LDA;            // A: row = 16*mw + nl
    const int bColBase = (128 * nh + nl) * LDB + 16 * lh; // B: n = 128*nh+16t+nl

    for (int kc = 0; kc < NDIM; kc += KC) {
        __syncthreads();   // previous chunk's WMMAs done before overwrite

        // ---- stage A: gather 128 atom rows x 32 K, split fp32 -> f16 hi/lo
        #pragma unroll
        for (int i = 0; i < (MTILE * KC) / 512; ++i) {
            int idx = tid + i * 512;
            int m = idx >> 5, j = idx & 31;
            int row = sortedIdx[off + min(mStart + m, cnt - 1)];
            float xv = X[(size_t)row * NDIM + kc + j];
            _Float16 hi = (_Float16)xv;
            _Float16 lo = (_Float16)(xv - (float)hi);
            sAh[m * LDA + j] = hi;
            sAl[m * LDA + j] = lo;
        }
        // ---- stage B: W1[e][kc..kc+31][0..255], transposed into [n][k]
        #pragma unroll 4
        for (int i = 0; i < (KC * NDIM) / 512; ++i) {
            int idx = tid + i * 512;
            int k = idx >> 8, n = idx & 255;
            float wv = W1[(size_t)e * NDIM * NDIM + (size_t)(kc + k) * NDIM + n];
            _Float16 hi = (_Float16)wv;
            _Float16 lo = (_Float16)(wv - (float)hi);
            sBh[n * LDB + k] = hi;
            sBl[n * LDB + k] = lo;
        }
        __syncthreads();

        // ---- A fragments (16-bit 16x32 layout, ISA 7.12.2):
        // lanes 0-15: M=lane; K=2q(+16*lh), contiguous to match B
        HFrag ah, al;
        #pragma unroll
        for (int q = 0; q < 4; ++q) {
            int K = 2 * q + 16 * lh;
            ah.u[q] = *(const unsigned int*)&sAh[aRowBase + K];
            al.u[q] = *(const unsigned int*)&sAl[aRowBase + K];
        }
        #pragma unroll
        for (int q = 4; q < 8; ++q) {
            int K = 2 * q + 16 * lh;
            ah.u[q] = *(const unsigned int*)&sAh[aRowBase + K];
            al.u[q] = *(const unsigned int*)&sAl[aRowBase + K];
        }

        // ---- 8 N-tiles for this wave's half; keep one B frag live at a time
        #pragma unroll
        for (int t = 0; t < 8; ++t) {
            const int bb = bColBase + t * (16 * LDB);
            HFrag bh;
            #pragma unroll
            for (int q = 0; q < 8; ++q)
                bh.u[q] = *(const unsigned int*)&sBh[bb + 2 * q];
            c[t] = __builtin_amdgcn_wmma_f32_16x16x32_f16(
                       false, ah.v, false, bh.v, (short)0, c[t], false, false);
            c[t] = __builtin_amdgcn_wmma_f32_16x16x32_f16(
                       false, al.v, false, bh.v, (short)0, c[t], false, false);
            HFrag bl;
            #pragma unroll
            for (int q = 0; q < 8; ++q)
                bl.u[q] = *(const unsigned int*)&sBl[bb + 2 * q];
            c[t] = __builtin_amdgcn_wmma_f32_16x16x32_f16(
                       false, ah.v, false, bl.v, (short)0, c[t], false, false);
        }
    }

    // ---- fused epilogue: bias + SiLU + layer-2 dot, all in registers.
    // C layout: lane l holds col N=128*nh+16t+(l%16), rows M = v + 8*lh.
    float part[8];
    #pragma unroll
    for (int v = 0; v < 8; ++v) part[v] = 0.f;

    #pragma unroll
    for (int t = 0; t < 8; ++t) {
        const int n  = 128 * nh + t * 16 + nl;
        const float bb = sB1[n];
        const float ww = sW2[n];
        #pragma unroll
        for (int v = 0; v < 8; ++v) {
            float h = c[t][v] + bb;
            float s = h / (1.f + __expf(-h));   // SiLU
            part[v] += s * ww;
        }
    }
    // reduce over 16 lanes sharing the same rows (xor masks stay within half)
    #pragma unroll
    for (int v = 0; v < 8; ++v) {
        part[v] += __shfl_xor(part[v], 1, 32);
        part[v] += __shfl_xor(part[v], 2, 32);
        part[v] += __shfl_xor(part[v], 4, 32);
        part[v] += __shfl_xor(part[v], 8, 32);
    }

    if (nl == 0) {
        #pragma unroll
        for (int v = 0; v < 8; ++v)
            sPart[nh * MTILE + 16 * mw + 8 * lh + v] = part[v];
    }
    __syncthreads();

    // combine the two N-halves and store one float per atom
    if (tid < MTILE) {
        int m = mStart + tid;
        if (m < cnt)
            out[sortedIdx[off + m]] = sPart[tid] + sPart[MTILE + tid] + b2[e];
    }
}

// ------------------------------- launcher ----------------------------------

extern "C" void kernel_launch(void* const* d_in, const int* in_sizes, int n_in,
                              void* d_out, int out_size, void* d_ws, size_t ws_size,
                              hipStream_t stream) {
    const float* X       = (const float*)d_in[0];
    const int*   species = (const int*)d_in[1];     // harness: integer -> int32
    const float* W1      = (const float*)d_in[2];
    const float* b1      = (const float*)d_in[3];
    const float* W2      = (const float*)d_in[4];
    const float* b2      = (const float*)d_in[5];
    float*       out     = (float*)d_out;

    const int nAtoms = in_sizes[1];
    const int nElem  = in_sizes[5];

    int* ws        = (int*)d_ws;
    int* counts    = ws;
    int* cursor    = ws + 16;
    int* offsets   = ws + 32;
    int* sortedIdx = ws + 48;

    erm_init<<<1, 64, 0, stream>>>(ws);
    erm_count<<<(nAtoms + 255) / 256, 256, 0, stream>>>(species, counts, nAtoms);
    erm_scan<<<1, 32, 0, stream>>>(ws, nElem);
    erm_scatter<<<(nAtoms + 255) / 256, 256, 0, stream>>>(species, cursor, sortedIdx, nAtoms);

    const int maxTiles = (nAtoms + MTILE - 1) / MTILE;  // over-provision; early-exit
    dim3 grid(maxTiles, nElem);
    erm_mlp<<<grid, 512, 0, stream>>>(X, W1, b1, W2, b2,
                                      counts, offsets, sortedIdx, out);
}
